// MyModel_C_4879082848679
// MI455X (gfx1250) — matrix operand order repaired and checked by reading, not compiled, not run on hardware
//
#include <hip/hip_runtime.h>
#include <hip/hip_bf16.h>
#include <math.h>

typedef __attribute__((ext_vector_type(16))) _Float16 v16h;
typedef __attribute__((ext_vector_type(8)))  _Float16 v8h;
typedef __attribute__((ext_vector_type(8)))  float    v8f;

__device__ inline v16h cat8(v8h a, v8h b) {
  return __builtin_shufflevector(a, b, 0,1,2,3,4,5,6,7,8,9,10,11,12,13,14,15);
}

__device__ inline v8f wmma_f16(v16h a, v16h b, v8f c) {
  // D = A(16x32 f16) * B(32x16 f16) + C(16x16 f32)
  return __builtin_amdgcn_wmma_f32_16x16x32_f16(false, a, false, b, (short)0, c, false, false);
}

// Load a pre-swizzled B fragment: frag base -> lane reads 16 contiguous halves.
__device__ inline v16h load_bfrag(const _Float16* fragBase, int lane) {
  const v8h* p = (const v8h*)(fragBase + lane * 16);
  return cat8(p[0], p[1]);
}

// Float atomic max/min via integer atomics (IEEE ordering trick); init -inf/+inf.
__device__ inline void atomicMaxF(float* addr, float value) {
  if (value >= 0.0f) atomicMax((int*)addr, __float_as_int(value));
  else               atomicMin((unsigned int*)addr, __float_as_uint(value));
}
__device__ inline void atomicMinF(float* addr, float value) {
  if (value >= 0.0f) atomicMin((int*)addr, __float_as_int(value));
  else               atomicMax((unsigned int*)addr, __float_as_uint(value));
}

// ---------------- embedding: xh = (nf @ w + b) as f16 [n,64] ----------------
__global__ void embed_kernel(const float* __restrict__ nf, const float* __restrict__ w,
                             const float* __restrict__ b, _Float16* __restrict__ xh,
                             int n, int F) {
  int idx = blockIdx.x * blockDim.x + threadIdx.x;
  if (idx >= n * 64) return;
  int node = idx >> 6, col = idx & 63;
  float s = b[col];
  for (int k = 0; k < F; ++k) s += nf[node * F + k] * w[k * 64 + col];
  xh[idx] = (_Float16)s;
}

// -------- swizzle f32 weight [K][ld] (cols colOff..colOff+N) into B-frag f16 --------
// out[((kt*NT+nt)*32 + lane)*16 + e] = W[kt*32 + (lane>>4)*8 + (e&7) + (e>>3)*16][colOff + nt*16 + (lane&15)]
__global__ void swz_kernel(const float* __restrict__ W, _Float16* __restrict__ out,
                           int K, int N, int ld, int colOff) {
  int t = blockIdx.x * blockDim.x + threadIdx.x;
  if (t >= K * N) return;
  int e = t & 15, lane = (t >> 4) & 31, fid = t >> 9;
  int NT = N >> 4;
  int nt = fid % NT, kt = fid / NT;
  int k = kt * 32 + (lane >> 4) * 8 + (e & 7) + (e >> 3) * 16;
  int n = nt * 16 + (lane & 15);
  out[t] = (_Float16)W[(size_t)k * ld + colOff + n];
}

// w2 [128][257] -> padded B-frag buffer with NT=17 tiles:
//   nt 0..15 : feature cols 1..256
//   nt 16    : gate column (w2[:,0]) in local col 0, zeros elsewhere
__global__ void swz_w2pad_kernel(const float* __restrict__ W, _Float16* __restrict__ out) {
  int t = blockIdx.x * blockDim.x + threadIdx.x;
  if (t >= 128 * 272) return;
  int e = t & 15, lane = (t >> 4) & 31, fid = t >> 9;
  int nt = fid % 17, kt = fid / 17;
  int k = kt * 32 + (lane >> 4) * 8 + (e & 7) + (e >> 3) * 16;
  int nl = lane & 15;
  float v;
  if (nt == 16) v = (nl == 0) ? W[(size_t)k * 257] : 0.0f;
  else          v = W[(size_t)k * 257 + 1 + nt * 16 + nl];
  out[t] = (_Float16)v;
}

// agg layout per node: [sum(64) | max(64) | min(64) | sum(64)]
__global__ void init_kernel(float* __restrict__ agg, int* __restrict__ cnt, int n) {
  int idx = blockIdx.x * blockDim.x + threadIdx.x;
  if (idx >= n * 256) return;
  int col = idx & 255, sec = col >> 6;
  agg[idx] = (sec == 1) ? -__builtin_inff() : (sec == 2) ? __builtin_inff() : 0.0f;
  if (col == 0) cnt[idx >> 8] = 0;
}

// ---------------- edge kernel: one 16-edge tile per wave ----------------
__global__ __launch_bounds__(128) void edge_kernel(
    const _Float16* __restrict__ srcH, const _Float16* __restrict__ dstH,
    const int* __restrict__ srcIdx, const int* __restrict__ dstIdx,
    int E, int nTiles,
    const _Float16* __restrict__ gW1, const _Float16* __restrict__ gW2,
    const float* __restrict__ gB1, const float* __restrict__ gB2,
    float* __restrict__ agg, int* __restrict__ cnt) {
  __shared__ __align__(16) _Float16 sW1[128 * 128];   // 32 KB
  __shared__ __align__(16) _Float16 sW2[128 * 272];   // 68 KB (16 feature tiles + gate tile)
  __shared__ float sB1[128];
  __shared__ float sB2[257];
  __shared__ __align__(16) _Float16 sH[4][16 * 128];  // per-wave relayout buffer

  int tid = threadIdx.x;
  { // cooperative weight fill (as 32-bit copies)
    const int* s1 = (const int*)gW1; int* d1 = (int*)sW1;
    for (int i = tid; i < 128 * 128 / 2; i += 128) d1[i] = s1[i];
    const int* s2 = (const int*)gW2; int* d2 = (int*)sW2;
    for (int i = tid; i < 128 * 272 / 2; i += 128) d2[i] = s2[i];
    if (tid < 128) sB1[tid] = gB1[tid];
    for (int i = tid; i < 257; i += 128) sB2[i] = gB2[i];
  }
  __syncthreads();

  int wave = tid >> 5, lane = tid & 31;
  int hi = lane >> 4, lo = lane & 15;
  _Float16* hbuf = sH[wave];

  for (int tile = blockIdx.x * 4 + wave; tile < nTiles; tile += gridDim.x * 4) {
    int e0 = tile * 16 + lo;
    int e = e0 < E ? e0 : E - 1;
    int si = srcIdx[e];
    int di = dstIdx[e];
    if (hi == 0 && e0 < E) atomicAdd(&cnt[di], 1);

    // ---- A fragments of xe = [src_x | dst_x], 16x128, f16 ----
    const _Float16* sr = srcH + (size_t)si * 64;
    const _Float16* dr = dstH + (size_t)di * 64;
    v16h a[4];
    a[0] = cat8(*(const v8h*)(sr + hi * 8),      *(const v8h*)(sr + 16 + hi * 8));
    a[1] = cat8(*(const v8h*)(sr + 32 + hi * 8), *(const v8h*)(sr + 48 + hi * 8));
    a[2] = cat8(*(const v8h*)(dr + hi * 8),      *(const v8h*)(dr + 16 + hi * 8));
    a[3] = cat8(*(const v8h*)(dr + 32 + hi * 8), *(const v8h*)(dr + 48 + hi * 8));

    // ---- layer 1: h = relu(xe @ W1 + b1), written as f16 into hbuf[16][128] ----
    #pragma unroll
    for (int nt = 0; nt < 8; ++nt) {
      v8f acc = {};
      #pragma unroll
      for (int kk = 0; kk < 4; ++kk)
        acc = wmma_f16(a[kk], load_bfrag(sW1 + ((kk * 8 + nt) * 32) * 16, lane), acc);
      int col = nt * 16 + lo;
      float bb = sB1[col];
      #pragma unroll
      for (int r = 0; r < 8; ++r) {
        float v = acc[r] + bb;
        v = v > 0.0f ? v : 0.0f;
        hbuf[(r + hi * 8) * 128 + col] = (_Float16)v;
      }
    }
    asm volatile("s_wait_dscnt 0" ::: "memory");
    __builtin_amdgcn_wave_barrier();

    // ---- A fragments of h ----
    v16h ha[4];
    #pragma unroll
    for (int kk = 0; kk < 4; ++kk) {
      const _Float16* rb = hbuf + lo * 128 + kk * 32;
      ha[kk] = cat8(*(const v8h*)(rb + hi * 8), *(const v8h*)(rb + 16 + hi * 8));
    }

    // ---- sigmoid gate via WMMA on the padded gate tile (nt=16) ----
    float kg[8];
    {
      v8f gacc = {};
      #pragma unroll
      for (int kk = 0; kk < 4; ++kk)
        gacc = wmma_f16(ha[kk], load_bfrag(sW2 + ((kk * 17 + 16) * 32) * 16, lane), gacc);
      float gb = sB2[0];
      #pragma unroll
      for (int r = 0; r < 8; ++r) {
        // column 0 of the C layout lives in lanes 0 (rows 0..7) and 16 (rows 8..15)
        float sg = 1.0f / (1.0f + __expf(-(gacc[r] + gb)));
        kg[r] = __shfl(sg, hi * 16);
      }
    }

    int dstr[8]; int evr[8];
    #pragma unroll
    for (int r = 0; r < 8; ++r) {
      dstr[r] = __shfl(di, r + hi * 8);
      evr[r]  = (tile * 16 + r + hi * 8) < E;
    }

    // ---- layer 2 features (cols 1..256 of w2), gated, scattered reductions ----
    #pragma unroll
    for (int nt = 0; nt < 16; ++nt) {
      v8f acc = {};
      #pragma unroll
      for (int kk = 0; kk < 4; ++kk)
        acc = wmma_f16(ha[kk], load_bfrag(sW2 + ((kk * 17 + nt) * 32) * 16, lane), acc);
      int col = nt * 16 + lo;          // 0..255
      float bb = sB2[1 + col];
      const int sec = nt >> 2;         // 0:sum 1:max 2:min 3:sum
      #pragma unroll
      for (int r = 0; r < 8; ++r) {
        if (!evr[r]) continue;
        float f = (acc[r] + bb) * kg[r];
        float* ap = agg + (size_t)dstr[r] * 256 + col;
        if (sec == 0 || sec == 3)      atomicAdd(ap, f);
        else if (sec == 1)             atomicMaxF(ap, f);
        else                           atomicMinF(ap, f);
      }
    }
    __builtin_amdgcn_wave_barrier();
  }
}

// ---------------- node kernel: one 16-node tile per wave ----------------
__global__ __launch_bounds__(128) void node_kernel(
    const _Float16* __restrict__ xh, const float* __restrict__ agg,
    const int* __restrict__ cnt,
    const _Float16* __restrict__ gRW, const float* __restrict__ gRB,
    const _Float16* __restrict__ gW1, const float* __restrict__ gB1,
    const _Float16* __restrict__ gW2, const float* __restrict__ gB2,
    float* __restrict__ out, int n, int nTiles) {
  __shared__ __align__(16) _Float16 sRW[320 * 64];  // 40 KB
  __shared__ __align__(16) _Float16 sW1[128 * 64];  // 16 KB
  __shared__ __align__(16) _Float16 sW2[64 * 64];   //  8 KB
  __shared__ float sRB[64], sB1[64], sB2[64];
  __shared__ __align__(16) _Float16 sY[4][16 * 128];
  __shared__ __align__(16) _Float16 sH2[4][16 * 64];

  int tid = threadIdx.x;
  {
    const int* s = (const int*)gRW; int* d = (int*)sRW;
    for (int i = tid; i < 320 * 64 / 2; i += 128) d[i] = s[i];
    s = (const int*)gW1; d = (int*)sW1;
    for (int i = tid; i < 128 * 64 / 2; i += 128) d[i] = s[i];
    s = (const int*)gW2; d = (int*)sW2;
    for (int i = tid; i < 64 * 64 / 2; i += 128) d[i] = s[i];
    if (tid < 64) { sRB[tid] = gRB[tid]; sB1[tid] = gB1[tid]; sB2[tid] = gB2[tid]; }
  }
  __syncthreads();

  int wave = tid >> 5, lane = tid & 31;
  int hi = lane >> 4, lo = lane & 15;
  _Float16* ybuf = sY[wave];
  _Float16* h2   = sH2[wave];

  for (int tile = blockIdx.x * 4 + wave; tile < nTiles; tile += gridDim.x * 4) {
    int node0 = tile * 16 + lo;
    int node = node0 < n ? node0 : n - 1;
    const _Float16* xr = xh + (size_t)node * 64;

    // A fragments of z = [x(64) | agg_fixed(256)], 16x320
    v16h az[10];
    az[0] = cat8(*(const v8h*)(xr + hi * 8),      *(const v8h*)(xr + 16 + hi * 8));
    az[1] = cat8(*(const v8h*)(xr + 32 + hi * 8), *(const v8h*)(xr + 48 + hi * 8));
    bool hasE = cnt[node] > 0;
    const float* ar = agg + (size_t)node * 256;
    #pragma unroll
    for (int kk = 2; kk < 10; ++kk) {
      int base = (kk - 2) * 32;
      int c0 = base + hi * 8, c1 = base + 16 + hi * 8;
      float4 f0 = *(const float4*)(ar + c0);
      float4 f1 = *(const float4*)(ar + c0 + 4);
      float4 g0 = *(const float4*)(ar + c1);
      float4 g1 = *(const float4*)(ar + c1 + 4);
      bool z0 = (!hasE) && (c0 >= 64 && c0 < 192);   // max/min sections -> 0 when no edges
      bool z1 = (!hasE) && (c1 >= 64 && c1 < 192);
      float tv[16] = {f0.x,f0.y,f0.z,f0.w,f1.x,f1.y,f1.z,f1.w,
                      g0.x,g0.y,g0.z,g0.w,g1.x,g1.y,g1.z,g1.w};
      v16h t;
      #pragma unroll
      for (int q = 0; q < 16; ++q) {
        float v = tv[q];
        if ((q < 8 && z0) || (q >= 8 && z1)) v = 0.0f;
        t[q] = (_Float16)v;
      }
      az[kk] = t;
    }

    // y[:,0:64] = x (for the output MLP input)
    {
      const v8h* s = (const v8h*)(xr + hi * 32);
      v8h* d = (v8h*)(ybuf + lo * 128 + hi * 32);
      d[0] = s[0]; d[1] = s[1]; d[2] = s[2]; d[3] = s[3];
    }

    // new_x = z @ rw + rb  -> y[:,64:128]
    #pragma unroll
    for (int nt = 0; nt < 4; ++nt) {
      v8f acc = {};
      #pragma unroll
      for (int kk = 0; kk < 10; ++kk)
        acc = wmma_f16(az[kk], load_bfrag(sRW + ((kk * 4 + nt) * 32) * 16, lane), acc);
      int col = nt * 16 + lo;
      float bb = sRB[col];
      #pragma unroll
      for (int r = 0; r < 8; ++r)
        ybuf[(r + hi * 8) * 128 + 64 + col] = (_Float16)(acc[r] + bb);
    }
    asm volatile("s_wait_dscnt 0" ::: "memory");
    __builtin_amdgcn_wave_barrier();

    // h2 = relu(y @ w1 + b1)
    v16h ya[4];
    #pragma unroll
    for (int kk = 0; kk < 4; ++kk) {
      const _Float16* rb = ybuf + lo * 128 + kk * 32;
      ya[kk] = cat8(*(const v8h*)(rb + hi * 8), *(const v8h*)(rb + 16 + hi * 8));
    }
    #pragma unroll
    for (int nt = 0; nt < 4; ++nt) {
      v8f acc = {};
      #pragma unroll
      for (int kk = 0; kk < 4; ++kk)
        acc = wmma_f16(ya[kk], load_bfrag(sW1 + ((kk * 4 + nt) * 32) * 16, lane), acc);
      int col = nt * 16 + lo;
      float bb = sB1[col];
      #pragma unroll
      for (int r = 0; r < 8; ++r) {
        float v = acc[r] + bb;
        v = v > 0.0f ? v : 0.0f;
        h2[(r + hi * 8) * 64 + col] = (_Float16)v;
      }
    }
    asm volatile("s_wait_dscnt 0" ::: "memory");
    __builtin_amdgcn_wave_barrier();

    // out = h2 @ w2 + b2
    v16h h2a[2];
    #pragma unroll
    for (int kk = 0; kk < 2; ++kk) {
      const _Float16* rb = h2 + lo * 64 + kk * 32;
      h2a[kk] = cat8(*(const v8h*)(rb + hi * 8), *(const v8h*)(rb + 16 + hi * 8));
    }
    #pragma unroll
    for (int nt = 0; nt < 4; ++nt) {
      v8f acc = {};
      #pragma unroll
      for (int kk = 0; kk < 2; ++kk)
        acc = wmma_f16(h2a[kk], load_bfrag(sW2 + ((kk * 4 + nt) * 32) * 16, lane), acc);
      int col = nt * 16 + lo;
      float bb = sB2[col];
      #pragma unroll
      for (int r = 0; r < 8; ++r) {
        int nrow = tile * 16 + r + hi * 8;
        if (nrow < n) out[(size_t)nrow * 64 + col] = acc[r] + bb;
      }
    }
    __builtin_amdgcn_wave_barrier();
  }
}

extern "C" void kernel_launch(void* const* d_in, const int* in_sizes, int n_in,
                              void* d_out, int out_size, void* d_ws, size_t ws_size,
                              hipStream_t stream) {
  const float* nf_gc  = (const float*)d_in[0];
  const float* nf_gs  = (const float*)d_in[1];
  const float* w_gc   = (const float*)d_in[2];
  const float* b_gc   = (const float*)d_in[3];
  const float* w_gs   = (const float*)d_in[4];
  const float* b_gs   = (const float*)d_in[5];
  const float* s2c_w1 = (const float*)d_in[6];
  const float* s2c_b1 = (const float*)d_in[7];
  const float* s2c_w2 = (const float*)d_in[8];
  const float* s2c_b2 = (const float*)d_in[9];
  const float* s2c_rw = (const float*)d_in[10];
  const float* s2c_rb = (const float*)d_in[11];
  const float* c2s_w1 = (const float*)d_in[12];
  const float* c2s_b1 = (const float*)d_in[13];
  const float* c2s_w2 = (const float*)d_in[14];
  const float* c2s_b2 = (const float*)d_in[15];
  const float* c2s_rw = (const float*)d_in[16];
  const float* c2s_rb = (const float*)d_in[17];
  const float* gc_w1  = (const float*)d_in[18];
  const float* gc_b1  = (const float*)d_in[19];
  const float* gc_w2  = (const float*)d_in[20];
  const float* gc_b2  = (const float*)d_in[21];
  const float* gs_w1  = (const float*)d_in[22];
  const float* gs_b1  = (const float*)d_in[23];
  const float* gs_w2  = (const float*)d_in[24];
  const float* gs_b2  = (const float*)d_in[25];
  const int* s2c_src  = (const int*)d_in[26];
  const int* s2c_dst  = (const int*)d_in[27];
  const int* c2s_src  = (const int*)d_in[28];
  const int* c2s_dst  = (const int*)d_in[29];
  float* out = (float*)d_out;

  const int NGC = in_sizes[0] / 10;
  const int NGS = in_sizes[1] / 8;
  const int E   = in_sizes[26];

  char* ws = (char*)d_ws;
  size_t off = 0;
  auto take = [&](size_t bytes) -> char* {
    char* p = ws + off;
    off = (off + bytes + 255) & ~(size_t)255;
    return p;
  };
  _Float16* gcH  = (_Float16*)take((size_t)NGC * 64 * 2);
  _Float16* gsH  = (_Float16*)take((size_t)NGS * 64 * 2);
  float*    aggC = (float*)take((size_t)NGC * 256 * 4);
  float*    aggS = (float*)take((size_t)NGS * 256 * 4);
  int*      cntC = (int*)take((size_t)NGC * 4);
  int*      cntS = (int*)take((size_t)NGS * 4);
  _Float16* s2cW1s = (_Float16*)take(128 * 128 * 2);
  _Float16* s2cW2s = (_Float16*)take(128 * 272 * 2);  // padded: 16 feature tiles + gate tile
  _Float16* c2sW1s = (_Float16*)take(128 * 128 * 2);
  _Float16* c2sW2s = (_Float16*)take(128 * 272 * 2);
  _Float16* rwCs = (_Float16*)take(320 * 64 * 2);
  _Float16* rwSs = (_Float16*)take(320 * 64 * 2);
  _Float16* w1Cs = (_Float16*)take(128 * 64 * 2);
  _Float16* w1Ss = (_Float16*)take(128 * 64 * 2);
  _Float16* w2Cs = (_Float16*)take(64 * 64 * 2);
  _Float16* w2Ss = (_Float16*)take(64 * 64 * 2);
  if (off > ws_size) return;

  // 1) embeddings (f16)
  embed_kernel<<<(NGC * 64 + 255) / 256, 256, 0, stream>>>(nf_gc, w_gc, b_gc, gcH, NGC, 10);
  embed_kernel<<<(NGS * 64 + 255) / 256, 256, 0, stream>>>(nf_gs, w_gs, b_gs, gsH, NGS, 8);

  // 2) weight swizzles to B-fragment layout (f16)
  auto swz = [&](const float* W, _Float16* o, int K, int N, int ld, int colOff) {
    swz_kernel<<<(K * N + 255) / 256, 256, 0, stream>>>(W, o, K, N, ld, colOff);
  };
  swz(s2c_w1, s2cW1s, 128, 128, 128, 0);
  swz(c2s_w1, c2sW1s, 128, 128, 128, 0);
  swz_w2pad_kernel<<<(128 * 272 + 255) / 256, 256, 0, stream>>>(s2c_w2, s2cW2s);
  swz_w2pad_kernel<<<(128 * 272 + 255) / 256, 256, 0, stream>>>(c2s_w2, c2sW2s);
  swz(s2c_rw, rwCs, 320, 64, 64, 0);
  swz(c2s_rw, rwSs, 320, 64, 64, 0);
  swz(gc_w1, w1Cs, 128, 64, 64, 0);
  swz(gs_w1, w1Ss, 128, 64, 64, 0);
  swz(gc_w2, w2Cs, 64, 64, 64, 0);
  swz(gs_w2, w2Ss, 64, 64, 64, 0);

  // 3) init aggregation buffers (sum/sum -> 0, max -> -inf, min -> +inf)
  init_kernel<<<(NGC * 256 + 255) / 256, 256, 0, stream>>>(aggC, cntC, NGC);
  init_kernel<<<(NGS * 256 + 255) / 256, 256, 0, stream>>>(aggS, cntS, NGS);

  // 4) edge MLPs + segment reductions
  const int nTilesE = (E + 15) / 16;
  edge_kernel<<<512, 128, 0, stream>>>(gsH, gcH, s2c_src, s2c_dst, E, nTilesE,
                                       s2cW1s, s2cW2s, s2c_b1, s2c_b2,
                                       aggC, cntC);
  edge_kernel<<<512, 128, 0, stream>>>(gcH, gsH, c2s_src, c2s_dst, E, nTilesE,
                                       c2sW1s, c2sW2s, c2s_b1, c2s_b2,
                                       aggS, cntS);

  // 5) node reduce-linear + output MLPs
  const int nTilesC = (NGC + 15) / 16;
  const int nTilesS = (NGS + 15) / 16;
  node_kernel<<<512, 128, 0, stream>>>(gcH, aggC, cntC, rwCs, s2c_rb,
                                       w1Cs, gc_b1, w2Cs, gc_b2,
                                       out, NGC, nTilesC);
  node_kernel<<<512, 128, 0, stream>>>(gsH, aggS, cntS, rwSs, c2s_rb,
                                       w1Ss, gs_b1, w2Ss, gs_b2,
                                       out + (size_t)NGC * 64, NGS, nTilesS);
}
